// SAM_21277267985157
// MI455X (gfx1250) — compile-verified, matrix operand order checked by reading, not run on hardware
//
#include <hip/hip_runtime.h>
#include <hip/hip_bf16.h>

// ---------------------------------------------------------------------------
// Self-attention (SAGAN-style) fused flash-attention implementation, gfx1250.
//   B=4, C=256, N=4096, CK=32.
//   - projections -> bf16 workspace (kT/qT as [N,32] K-major, v as [C,N]);
//     log2(e) folded into the k projection so softmax uses bare v_exp_f32.
//   - fused S^T = Q K^T -> lane-local online softmax -> PV, all WMMA bf16.
//   - S^T orientation makes softmax per-lane: P^T is repacked to the WMMA
//     B-operand layout entirely in registers (no LDS, 2 shuffles/iteration);
//     packing is 8x v_perm_b32 (truncating f32->bf16, within bf16 noise).
//   - 2 waves / block, 128 channels per wave: 2 S + 8 PV WMMAs per j-block
//     (90% of block WMMAs are unique work), accumulator rescale skipped via
//     EXECZ branch once the running max stabilizes.
// ---------------------------------------------------------------------------

#define BB   4
#define CCH  256
#define CK   32
#define NPIX 4096
#define LOG2E 1.4426950408889634f

typedef __attribute__((ext_vector_type(16))) __bf16        v16bf;
typedef __attribute__((ext_vector_type(8)))  float         v8f;
typedef __attribute__((ext_vector_type(8)))  unsigned int  v8u;
typedef __attribute__((ext_vector_type(4)))  unsigned int  v4u;

__device__ __forceinline__ unsigned int f2bf_u(float f) {
    unsigned int u = __float_as_uint(f);
    u += 0x7FFFu + ((u >> 16) & 1u);      // round-to-nearest-even
    return u >> 16;
}
__device__ __forceinline__ unsigned short f2bf(float f) {
    return (unsigned short)f2bf_u(f);
}

#if defined(__has_builtin) && __has_builtin(__builtin_amdgcn_exp2f)
__device__ __forceinline__ float fast_exp2(float x) { return __builtin_amdgcn_exp2f(x); }
#else
__device__ __forceinline__ float fast_exp2(float x) { return exp2f(x); }
#endif

// Pack two f32 into one dword of two bf16 (truncation) with one v_perm_b32:
// bytes {lo.3, lo.2} -> result[15:0], {hi.3, hi.2} -> result[31:16].
__device__ __forceinline__ unsigned int pack_bf16_trunc(float lo, float hi) {
    return __builtin_amdgcn_perm(__float_as_uint(hi), __float_as_uint(lo),
                                 0x07060302u);
}

// Load one 16x32 bf16 WMMA operand fragment (A-layout, or B-layout of the
// transposed matrix -- identical access pattern).  `p` already points at
// element (row = lane%16, k = kb); reads elements [kb..kb+7] and [kb+16..kb+23]
// as two 16-byte chunks.
__device__ __forceinline__ v16bf ldfrag(const unsigned short* p) {
    v4u lo = *(const v4u*)(p);
    v4u hi = *(const v4u*)(p + 16);
    v8u r;
    r[0] = lo[0]; r[1] = lo[1]; r[2] = lo[2]; r[3] = lo[3];
    r[4] = hi[0]; r[5] = hi[1]; r[6] = hi[2]; r[7] = hi[3];
    return __builtin_bit_cast(v16bf, r);
}

// ---------------------------------------------------------------------------
// Kernel 1: k/q projections.  kT/qT stored as [B][N][32] bf16 (K-major rows).
// k is pre-scaled by log2(e) so the attention kernel works in base-2 logits.
// ---------------------------------------------------------------------------
__global__ __launch_bounds__(256) void proj_kq(
    const float* __restrict__ x,
    const float* __restrict__ kW, const float* __restrict__ kb,
    const float* __restrict__ qW, const float* __restrict__ qb,
    unsigned short* __restrict__ kT, unsigned short* __restrict__ qT)
{
    __shared__ float sKW[CK * CCH];
    __shared__ float sQW[CK * CCH];
    const int b = blockIdx.x;
    const int n = blockIdx.y * 256 + threadIdx.x;

    for (int i = threadIdx.x; i < CK * CCH; i += 256) {
        sKW[i] = kW[i] * LOG2E;          // fold log2(e) into k
        sQW[i] = qW[i];
    }
    __syncthreads();

    const float* xb = x + (size_t)b * CCH * NPIX + n;
    float ak[CK], aq[CK];
#pragma unroll
    for (int k = 0; k < CK; ++k) { ak[k] = 0.f; aq[k] = 0.f; }

    for (int c = 0; c < CCH; ++c) {
        float xv = xb[(size_t)c * NPIX];
#pragma unroll
        for (int k = 0; k < CK; ++k) {
            ak[k] += sKW[k * CCH + c] * xv;
            aq[k] += sQW[k * CCH + c] * xv;
        }
    }

    unsigned short* ko = kT + ((size_t)b * NPIX + n) * CK;
    unsigned short* qo = qT + ((size_t)b * NPIX + n) * CK;
#pragma unroll
    for (int k = 0; k < CK; ++k) {
        ko[k] = f2bf(ak[k] + kb[k] * LOG2E);
        qo[k] = f2bf(aq[k] + qb[k]);
    }
}

// ---------------------------------------------------------------------------
// Kernel 2: v projection.  v stored as [B][C][N] bf16 (j contiguous -> K-major
// rows for the PV GEMM A operand).
// ---------------------------------------------------------------------------
__global__ void proj_v(
    const float* __restrict__ x,
    const float* __restrict__ vW, const float* __restrict__ vb,
    unsigned short* __restrict__ vB)
{
    const int b  = blockIdx.z;
    const int n  = blockIdx.x * 64 + threadIdx.x;
    const int c0 = blockIdx.y * 16 + threadIdx.y * 4;

    const float* xb = x + (size_t)b * CCH * NPIX + n;
    float acc[4] = {0.f, 0.f, 0.f, 0.f};

    for (int c = 0; c < CCH; ++c) {
        float xv = xb[(size_t)c * NPIX];
#pragma unroll
        for (int k = 0; k < 4; ++k)
            acc[k] += vW[(size_t)(c0 + k) * CCH + c] * xv;
    }
#pragma unroll
    for (int k = 0; k < 4; ++k)
        vB[((size_t)b * CCH + c0 + k) * NPIX + n] = f2bf(acc[k] + vb[c0 + k]);
}

// ---------------------------------------------------------------------------
// Kernel 3: fused attention.
//   Block = 16 query rows (i-tile) of one batch, 2 waves; each wave owns 128
//   output channels (8 PV tiles).  Per j-block of 32 keys:
//     T = S^T = Q_tile K_tile^T      (2 WMMAs, K=CK=32, base-2 logits)
//     lane-local online softmax      (max/sum over registers + 1 shfl_xor(16))
//     P^T packed to bf16 B-operand   (8x v_perm_b32, registers only)
//     O[c,i] += V[c,j] P^T[j,i]      (8 WMMAs, K=32, independent accumulators)
// ---------------------------------------------------------------------------
__global__ __launch_bounds__(64) void attn_fused(
    const unsigned short* __restrict__ kT,
    const unsigned short* __restrict__ qT,
    const unsigned short* __restrict__ vB,
    const float* __restrict__ x,
    const float* __restrict__ alpha_p,
    float* __restrict__ out)
{
    const int b     = blockIdx.y;
    const int iBase = blockIdx.x * 16;
    const int w     = threadIdx.x >> 5;         // wave -> channels [w*128, w*128+128)
    const int lane  = threadIdx.x & 31;
    const int lrow  = lane & 15;
    const int kb    = (lane & 16) ? 8 : 0;      // K sub-block per ISA 16-bit layout
    const int half8 = (lane & 16) ? 8 : 0;      // C/D layout row offset

    const unsigned short* kTb = kT + (size_t)b * NPIX * CK;
    const unsigned short* qTb = qT + (size_t)b * NPIX * CK;
    const unsigned short* vBb = vB + (size_t)b * CCH * NPIX;

    // B operand of the QK GEMM: K^T tile, columns i = iBase..iBase+15 (fixed).
    const v16bf bK = ldfrag(kTb + (size_t)(iBase + lrow) * CK + kb);

    v8f acc[8] = {};
    v8f zero = {};
    float mcol = -3e38f;     // running max of column i (lane-local)
    float lcol = 0.f;        // running sum of column i (lane-local)

    for (int jb = 0; jb < NPIX; jb += 32) {
        // ---- T = S^T for 32 key rows j ----------------------------------
        v16bf aQ0 = ldfrag(qTb + (size_t)(jb + lrow) * CK + kb);
        v16bf aQ1 = ldfrag(qTb + (size_t)(jb + 16 + lrow) * CK + kb);
        v8f T0 = __builtin_amdgcn_wmma_f32_16x16x32_bf16(false, aQ0, false, bK,
                                                         (short)0, zero, false, false);
        v8f T1 = __builtin_amdgcn_wmma_f32_16x16x32_bf16(false, aQ1, false, bK,
                                                         (short)0, zero, false, false);

        // speculative prefetch of next q rows (non-faulting, branch-free)
        __builtin_prefetch(qTb + (size_t)(jb + 32 + lrow) * CK, 0, 3);
        __builtin_prefetch(qTb + (size_t)(jb + 48 + lrow) * CK, 0, 3);

        // ---- lane-local online softmax over column i (base-2) -----------
        float mx = fmaxf(T0[0], T1[0]);
#pragma unroll
        for (int r = 1; r < 8; ++r) mx = fmaxf(mx, fmaxf(T0[r], T1[r]));
        mx = fmaxf(mx, __shfl_xor(mx, 16));          // combine lane halves
        const float mnew = fmaxf(mcol, mx);
        const float sc   = fast_exp2(mcol - mnew);

        float P0[8], P1[8];
        float ps = 0.f;
#pragma unroll
        for (int r = 0; r < 8; ++r) {
            P0[r] = fast_exp2(T0[r] - mnew);
            P1[r] = fast_exp2(T1[r] - mnew);
            ps += P0[r] + P1[r];
        }
        ps += __shfl_xor(ps, 16);                    // combine lane halves
        lcol = fmaf(lcol, sc, ps);
        mcol = mnew;

        // rescale only when the running max actually moved (EXECZ skip once
        // the max stabilizes; predication keeps sc==1 lanes untouched).
        if (sc < 1.0f) {
#pragma unroll
            for (int t = 0; t < 8; ++t)
#pragma unroll
                for (int r = 0; r < 8; ++r) acc[t][r] *= sc;
        }

        // ---- pack P^T into the B-operand layout (registers only) --------
        // VGPR v holds K = kb+2v, kb+2v+1 (from T0) and VGPR 4+v holds
        // K = kb+16+2v, kb+16+2v+1 (from T1) -- exactly our register layout.
        v8u pu;
#pragma unroll
        for (int v = 0; v < 4; ++v) {
            pu[v]     = pack_bf16_trunc(P0[2 * v], P0[2 * v + 1]);
            pu[4 + v] = pack_bf16_trunc(P1[2 * v], P1[2 * v + 1]);
        }
        const v16bf bP = __builtin_bit_cast(v16bf, pu);

        // ---- O[c,i] += V[c,j] * P[i,j]  (A = V tile, B = P^T) -----------
#pragma unroll
        for (int t = 0; t < 8; ++t) {
            const unsigned short* vr =
                vBb + (size_t)(w * 128 + t * 16 + lrow) * NPIX + jb + kb;
            __builtin_prefetch(vr + 32, 0, 3);
            v16bf aV = ldfrag(vr);
            acc[t] = __builtin_amdgcn_wmma_f32_16x16x32_bf16(false, aV, false, bP,
                                                             (short)0, acc[t],
                                                             false, false);
        }
    }

    // ---- normalize, blend with residual, store --------------------------
    const float inv   = 1.0f / lcol;                 // lane-local column sum
    const float alpha = alpha_p[0];
    const int   i     = iBase + lrow;
    const float* xb   = x   + (size_t)b * CCH * NPIX;
    float*       ob   = out + (size_t)b * CCH * NPIX;
#pragma unroll
    for (int t = 0; t < 8; ++t)
#pragma unroll
        for (int r = 0; r < 8; ++r) {
            const int c = w * 128 + t * 16 + r + half8;
            ob[(size_t)c * NPIX + i] =
                alpha * (acc[t][r] * inv) + xb[(size_t)c * NPIX + i];
        }
}

// ---------------------------------------------------------------------------
extern "C" void kernel_launch(void* const* d_in, const int* in_sizes, int n_in,
                              void* d_out, int out_size, void* d_ws, size_t ws_size,
                              hipStream_t stream) {
    const float* x   = (const float*)d_in[0];
    const float* kW  = (const float*)d_in[1];
    const float* kb  = (const float*)d_in[2];
    const float* qW  = (const float*)d_in[3];
    const float* qb  = (const float*)d_in[4];
    const float* vW  = (const float*)d_in[5];
    const float* vb  = (const float*)d_in[6];
    const float* alp = (const float*)d_in[7];
    float* out = (float*)d_out;

    unsigned short* kT = (unsigned short*)d_ws;                 // [B][N][32] bf16
    unsigned short* qT = kT + (size_t)BB * NPIX * CK;           // [B][N][32] bf16
    unsigned short* vB = qT + (size_t)BB * NPIX * CK;           // [B][C][N] bf16

    proj_kq<<<dim3(BB, NPIX / 256), dim3(256), 0, stream>>>(x, kW, kb, qW, qb, kT, qT);
    proj_v<<<dim3(NPIX / 64, CCH / 16, BB), dim3(64, 4), 0, stream>>>(x, vW, vb, vB);
    attn_fused<<<dim3(NPIX / 16, BB), dim3(64), 0, stream>>>(kT, qT, vB, x, alp, out);
}
